// FractalAttentionHead_25134148616685
// MI455X (gfx1250) — matrix operand order repaired and checked by reading, not compiled, not run on hardware
//
#include <hip/hip_runtime.h>

typedef __bf16 bf16;
typedef __bf16 v16bf __attribute__((ext_vector_type(16)));
typedef __bf16 v8bf  __attribute__((ext_vector_type(8)));
typedef float  v8f   __attribute__((ext_vector_type(8)));
typedef unsigned int v4u __attribute__((ext_vector_type(4)));
typedef int v4i_vs __attribute__((vector_size(16)));   // matches builtin's pointee type

#define DIM   1024
#define BATCH 8
#define SEQ   1024

// GEMM tiling
#define KC 64                       // K elements staged per chunk (2 wmma K-steps)
#define RS 144                      // LDS row stride bytes: 64*2 + 16 pad (16B aligned, conflict-free)
#define TILE_BYTES (128 * RS)       // one 128xKC bf16 tile  = 18432 B
#define BUF_BYTES  (4 * TILE_BYTES) // Ahi,Alo,Bhi,Blo       = 73728 B
#define SMEM_BYTES (2 * BUF_BYTES)  // double buffered       = 147456 B (<320KB WGP LDS)

#if __has_builtin(__builtin_amdgcn_global_load_async_to_lds_b128) && \
    __has_builtin(__builtin_amdgcn_s_wait_asynccnt)
#define HAVE_ASYNC 1
#else
#define HAVE_ASYNC 0
#endif

__device__ __forceinline__ void copy16_g2l(const bf16* g, char* l) {
#if HAVE_ASYNC
    __builtin_amdgcn_global_load_async_to_lds_b128(
        (__attribute__((address_space(1))) v4i_vs*)g,
        (__attribute__((address_space(3))) v4i_vs*)l, 0, 0);
#else
    *(v4u*)l = *(const v4u*)g;  // synchronous fallback
#endif
}

#if HAVE_ASYNC
#define WAIT_ASYNC(n) __builtin_amdgcn_s_wait_asynccnt(n)
#else
#define WAIT_ASYNC(n)
#endif

__device__ __forceinline__ void split2(float x, bf16& h, bf16& l) {
    h = (bf16)x;
    l = (bf16)(x - (float)h);
}

union FragBF {
    v16bf v;
    v8bf  h[2];
};

__device__ __forceinline__ v8f wmma_bf16(v16bf a, v16bf b, v8f c) {
    return __builtin_amdgcn_wmma_f32_16x16x32_bf16(false, a, false, b, (short)0, c, false, false);
}

// ---------------------------------------------------------------------------
// split fp32 array into bf16 hi/lo pair  (a = hi + lo, ~16 mantissa bits)
// ---------------------------------------------------------------------------
__global__ __launch_bounds__(256)
void split_kernel(const float* __restrict__ in, bf16* __restrict__ hi,
                  bf16* __restrict__ lo, unsigned int n) {
    const unsigned int i = blockIdx.x * 256u + threadIdx.x;
    if (i < n) split2(in[i], hi[i], lo[i]);
}

// ---------------------------------------------------------------------------
// WsumT[e][d] = sum_n w[n][d][e], emitted as bf16 hi/lo
// ---------------------------------------------------------------------------
__global__ __launch_bounds__(256)
void wsum_transpose_split_kernel(const float* __restrict__ w,
                                 bf16* __restrict__ hi, bf16* __restrict__ lo) {
    __shared__ float tile[32][33];
    const int bx = blockIdx.x * 32;  // e base
    const int by = blockIdx.y * 32;  // d base
    const int tx = threadIdx.x;      // 0..31
    const int ty = threadIdx.y;      // 0..7
#pragma unroll
    for (int i = 0; i < 32; i += 8) {
        const size_t off = (size_t)(by + ty + i) * DIM + bx + tx;
        tile[ty + i][tx] = w[off] + w[(size_t)DIM * DIM + off] + w[2 * (size_t)DIM * DIM + off];
    }
    __syncthreads();
#pragma unroll
    for (int i = 0; i < 32; i += 8) {
        const float s = tile[tx][ty + i];
        const size_t o = (size_t)(bx + ty + i) * DIM + by + tx;
        bf16 h, l;
        split2(s, h, l);
        hi[o] = h;
        lo[o] = l;
    }
}

// ---------------------------------------------------------------------------
// NT GEMM with bf16x3 fp32 emulation: C = A[M,K] * B[N,K]^T
//   acc += Ahi*Bhi + Ahi*Blo + Alo*Bhi   (fp32 accumulate, lo*lo dropped)
//   MODE 0: val = acc + bias[n]; n<1024 -> q hi/lo, n<2048 -> k hi/lo, else v f32
//   MODE 1: val = acc          -> t hi/lo
//   MODE 2: OutF = V + scale*acc, batched over blockIdx.z
// Block 128x128, 8 waves (wave tile 64x32 = 4x2 16x16 tiles), KC=64,
// double-buffered LDS fed by async global->LDS copies.
// ---------------------------------------------------------------------------
template <int MODE>
__global__ __launch_bounds__(256)
void gemm_bf16x3(const bf16* __restrict__ Ahi, const bf16* __restrict__ Alo, int lda,
                 const bf16* __restrict__ Bhi, const bf16* __restrict__ Blo, int ldb,
                 const float* __restrict__ bias,
                 bf16* __restrict__ OutHi0, bf16* __restrict__ OutLo0,
                 bf16* __restrict__ OutHi1, bf16* __restrict__ OutLo1,
                 float* __restrict__ OutF, const float* __restrict__ V, float scale,
                 int K, size_t zStrideA, size_t zStrideB, size_t zStrideC) {
    extern __shared__ char smem[];

    const int z = blockIdx.z;
    Ahi += (size_t)z * zStrideA;
    Alo += (size_t)z * zStrideA;
    Bhi += (size_t)z * zStrideB;
    Blo += (size_t)z * zStrideB;
    if (MODE == 2) {
        OutF += (size_t)z * zStrideC;
        V    += (size_t)z * zStrideC;
    }

    const int tid   = threadIdx.x;
    const int lane  = tid & 31;
    const int wave  = tid >> 5;
    const int waveM = wave & 1;   // 0..1 -> 64-row slab
    const int waveN = wave >> 1;  // 0..3 -> 32-col slab
    const int half  = lane >> 4;
    const int lcol  = lane & 15;

    const int blockM = blockIdx.y * 128;
    const int blockN = blockIdx.x * 128;

    // global->LDS mapping: per array, 4 passes of one b128 per thread
    const int grow = tid >> 3;         // 0..31
    const int gcol = (tid & 7) << 3;   // 0..56 step 8 (bf16 elements)

    v8f acc[4][2];
#pragma unroll
    for (int mi = 0; mi < 4; ++mi)
#pragma unroll
        for (int ni = 0; ni < 2; ++ni)
#pragma unroll
            for (int r = 0; r < 8; ++r) acc[mi][ni][r] = 0.0f;

    auto issue = [&](int kc, int buf) {
        char* base = smem + buf * BUF_BYTES;
#pragma unroll
        for (int p = 0; p < 4; ++p) {
            const int r = grow + p * 32;
            const size_t ga = (size_t)(blockM + r) * lda + kc + gcol;
            const size_t gb = (size_t)(blockN + r) * ldb + kc + gcol;
            const int lo = r * RS + gcol * 2;  // LDS byte offset within tile
            copy16_g2l(Ahi + ga, base + lo);
            copy16_g2l(Alo + ga, base + TILE_BYTES + lo);
            copy16_g2l(Bhi + gb, base + 2 * TILE_BYTES + lo);
            copy16_g2l(Blo + gb, base + 3 * TILE_BYTES + lo);
        }
    };

    auto compute = [&](int buf) {
        const char* base = smem + buf * BUF_BYTES;
#pragma unroll
        for (int k32 = 0; k32 < KC; k32 += 32) {
            // B fragments (32x16 bf16): lane = N col, VGPRs = 16 contiguous K @ 16*half
            FragBF bh[2], bl[2];
#pragma unroll
            for (int ni = 0; ni < 2; ++ni) {
                const char* bp =
                    base + 2 * TILE_BYTES + (waveN * 32 + ni * 16 + lcol) * RS + 2 * k32 + 32 * half;
                bh[ni].h[0] = *(const v8bf*)bp;
                bh[ni].h[1] = *(const v8bf*)(bp + 16);
                bl[ni].h[0] = *(const v8bf*)(bp + TILE_BYTES);
                bl[ni].h[1] = *(const v8bf*)(bp + TILE_BYTES + 16);
            }
#pragma unroll
            for (int mi = 0; mi < 4; ++mi) {
                // A fragment (16x32 bf16): lane = M row; VGPRs = 16 contiguous K @ 16*half
                const char* ap =
                    base + (waveM * 64 + mi * 16 + lcol) * RS + 2 * k32 + 32 * half;
                FragBF ah, al;
                ah.h[0] = *(const v8bf*)ap;
                ah.h[1] = *(const v8bf*)(ap + 16);
                al.h[0] = *(const v8bf*)(ap + TILE_BYTES);
                al.h[1] = *(const v8bf*)(ap + TILE_BYTES + 16);
#pragma unroll
                for (int ni = 0; ni < 2; ++ni) {
                    acc[mi][ni] = wmma_bf16(ah.v, bh[ni].v, acc[mi][ni]);
                    acc[mi][ni] = wmma_bf16(al.v, bh[ni].v, acc[mi][ni]);
                    acc[mi][ni] = wmma_bf16(ah.v, bl[ni].v, acc[mi][ni]);
                }
            }
        }
    };

    const int NCHUNK = K / KC;
    issue(0, 0);
    for (int i = 0; i < NCHUNK; ++i) {
        __syncthreads();  // all waves done with chunk i-1 => buffer (i+1)&1 reusable
        if (i + 1 < NCHUNK) {
            issue((i + 1) * KC, (i + 1) & 1);
            WAIT_ASYNC(16);  // in-order ASYNCcnt: chunk i's 16 loads have landed
        } else {
            WAIT_ASYNC(0);
        }
        __syncthreads();  // chunk i visible to every wave
        compute(i & 1);
    }

    // Epilogue: C 16x16 f32 layout -> VGPR r: M = r + 8*half, lane%16 -> N
#pragma unroll
    for (int mi = 0; mi < 4; ++mi) {
#pragma unroll
        for (int ni = 0; ni < 2; ++ni) {
            const int gn  = blockN + waveN * 32 + ni * 16 + lcol;
            const int gm0 = blockM + waveM * 64 + mi * 16 + half * 8;
#pragma unroll
            for (int r = 0; r < 8; ++r) {
                const int gm = gm0 + r;
                float val = acc[mi][ni][r];
                if (MODE == 0) {
                    val += bias[gn];
                    const int seg = gn >> 10;          // uniform per block (128 | 1024)
                    const size_t o = (size_t)gm * DIM + (gn & 1023);
                    bf16 h, l;
                    if (seg == 0) { split2(val, h, l); OutHi0[o] = h; OutLo0[o] = l; }
                    else if (seg == 1) { split2(val, h, l); OutHi1[o] = h; OutLo1[o] = l; }
                    else { OutF[o] = val; }
                } else if (MODE == 1) {
                    const size_t o = (size_t)gm * DIM + gn;
                    bf16 h, l;
                    split2(val, h, l);
                    OutHi0[o] = h;
                    OutLo0[o] = l;
                } else {
                    const size_t o = (size_t)gm * DIM + gn;
                    OutF[o] = V[o] + scale * val;
                }
            }
        }
    }
}

// ---------------------------------------------------------------------------
extern "C" void kernel_launch(void* const* d_in, const int* in_sizes, int n_in,
                              void* d_out, int out_size, void* d_ws, size_t ws_size,
                              hipStream_t stream) {
    const float* x    = (const float*)d_in[0];  // [8,1024,1024]
    const float* Wqkv = (const float*)d_in[1];  // [3072,1024]
    const float* bqkv = (const float*)d_in[2];  // [3072]
    const float* w    = (const float*)d_in[3];  // [3,1024,1024]
    float* out = (float*)d_out;                 // [8,1024,1024]

    const size_t MS = (size_t)BATCH * SEQ;      // 8192 rows
    char* p = (char*)d_ws;
    auto carve = [&](size_t bytes) -> char* {
        char* r = p;
        p += (bytes + 255) & ~(size_t)255;
        return r;
    };
    bf16* x_hi  = (bf16*)carve(MS * DIM * 2);
    bf16* x_lo  = (bf16*)carve(MS * DIM * 2);
    bf16* Wq_hi = (bf16*)carve((size_t)3 * DIM * DIM * 2);
    bf16* Wq_lo = (bf16*)carve((size_t)3 * DIM * DIM * 2);
    bf16* Ws_hi = (bf16*)carve((size_t)DIM * DIM * 2);
    bf16* Ws_lo = (bf16*)carve((size_t)DIM * DIM * 2);
    bf16* q_hi  = (bf16*)carve(MS * DIM * 2);
    bf16* q_lo  = (bf16*)carve(MS * DIM * 2);
    bf16* k_hi  = (bf16*)carve(MS * DIM * 2);
    bf16* k_lo  = (bf16*)carve(MS * DIM * 2);
    bf16* t_hi  = (bf16*)carve(MS * DIM * 2);
    bf16* t_lo  = (bf16*)carve(MS * DIM * 2);
    float* vf   = (float*)carve(MS * DIM * 4);

    // 1) precision split of the fp32 GEMM inputs (memory-bound, tiny)
    {
        const unsigned int nx = (unsigned int)(MS * DIM);
        split_kernel<<<dim3(nx / 256), 256, 0, stream>>>(x, x_hi, x_lo, nx);
        const unsigned int nw = (unsigned int)(3u * DIM * DIM);
        split_kernel<<<dim3(nw / 256), 256, 0, stream>>>(Wqkv, Wq_hi, Wq_lo, nw);
    }

    // 2) WsumT = (w[0]+w[1]+w[2])^T  -> bf16 hi/lo
    wsum_transpose_split_kernel<<<dim3(DIM / 32, DIM / 32), dim3(32, 8), 0, stream>>>(
        w, Ws_hi, Ws_lo);

    // 3) qkv = x @ Wqkv^T + b : M=8192, N=3072, K=1024 -> q,k (bf16 hi/lo) + v (f32)
    gemm_bf16x3<0><<<dim3(3 * DIM / 128, MS / 128, 1), 256, SMEM_BYTES, stream>>>(
        x_hi, x_lo, DIM, Wq_hi, Wq_lo, DIM, bqkv,
        q_hi, q_lo, k_hi, k_lo, vf, nullptr, 0.0f, DIM, 0, 0, 0);

    // 4) t = q @ Wsum : M=8192, N=1024, K=1024 -> t (bf16 hi/lo)
    gemm_bf16x3<1><<<dim3(DIM / 128, MS / 128, 1), 256, SMEM_BYTES, stream>>>(
        q_hi, q_lo, DIM, Ws_hi, Ws_lo, DIM, nullptr,
        t_hi, t_lo, nullptr, nullptr, nullptr, nullptr, 0.0f, DIM, 0, 0, 0);

    // 5) out_b = v_b + scale * (t_b @ k_b^T) : per batch M=N=K=1024
    const float scale = 0.03125f;  // 1/sqrt(1024)
    gemm_bf16x3<2><<<dim3(DIM / 128, SEQ / 128, BATCH), 256, SMEM_BYTES, stream>>>(
        t_hi, t_lo, DIM, k_hi, k_lo, DIM, nullptr,
        nullptr, nullptr, nullptr, nullptr, out, vf, scale, DIM,
        (size_t)SEQ * DIM, (size_t)SEQ * DIM, (size_t)SEQ * DIM);
}
